// Arch9GraphEncoder_67053029425456
// MI455X (gfx1250) — compile-verified, matrix operand order checked by reading, not run on hardware
//
#include <hip/hip_runtime.h>
#include <hip/hip_bf16.h>

// ---------------- model constants ----------------
#define HH      128
#define LL      4
#define NT      4096          // N_TOTAL
#define MM      4
#define SS      16384         // S
#define KK      12            // K
#define SKK     196608        // S*K
#define EINTRA  180224        // S*(K-1)
#define ECANON  32768         // N_TOTAL*8
#define BB      64
#define MAXD    32

typedef __attribute__((ext_vector_type(16))) _Float16 v16h;
typedef __attribute__((ext_vector_type(8)))  _Float16 h8;
typedef __attribute__((ext_vector_type(8)))  float    v8f;

union Frag { v16h v; h8 h[2]; };

// =====================================================================
//  WMMA GEMM:  Y[row, col] = act( sum_k X[xrow(row), k] * Wh[col, k] + bias[col] ) (+R)
//  X f32 rows gathered with stride xstep; Wh is pre-converted f16 [OC, 128].
//  Weight tile staged into LDS with global_load_async_to_lds_b128 (ASYNCcnt).
//  The LDS destination is passed as (unsigned)(size_t)&Wl[...] so the shared
//  array's address ESCAPES into the asm -> GlobalsModRef must assume the asm
//  writes it (fixes the undef-fold of the B operands seen previously).
//  grid.x = nrows/128, grid.y = OC/128; block = 256 (8 waves, 16 rows each).
// =====================================================================
__global__ __launch_bounds__(256) void gemm_wmma_kernel(
    const float* __restrict__ X, int xstep,
    const _Float16* __restrict__ Wh,   // [OC, 128] row-major f16
    const float* __restrict__ bias,    // [OC]
    const float* __restrict__ R,       // optional residual (ld = ldY) or nullptr
    float* __restrict__ Y, int ldY, int do_relu)
{
    __shared__ _Float16 Wl[128 * 136];          // +8 halves padding -> conflict-free
    const int tid = threadIdx.x;
    const int colbase = blockIdx.y * 128;

    // ---- async-stage this block's 128x128 f16 weight tile into LDS ----
    // 2048 chunks of 16B; per chunk: row o, 8-half segment seg.
#pragma unroll
    for (int it = 0; it < 8; ++it) {
        const int c = tid + it * 256;
        const int o = c >> 4, seg = c & 15;
        const _Float16* gsrc = Wh + (size_t)(colbase + o) * HH + seg * 8;
        // low 32 bits of the generic (flat) address of an LDS location == DS byte offset
        const unsigned ldso = (unsigned)(size_t)(&Wl[o * 136 + seg * 8]);
        asm volatile("global_load_async_to_lds_b128 %0, %1, off"
                     :: "v"(ldso), "v"(gsrc) : "memory");
    }
    asm volatile("s_wait_asynccnt 0x0" ::: "memory");
    __syncthreads();

    const int wave = tid >> 5, lane = tid & 31;
    const int half = lane >> 4, m = lane & 15;
    const int rowbase = blockIdx.x * 128 + wave * 16;
    const float* __restrict__ Xrow = X + (size_t)(rowbase + m) * xstep * HH;
    const float4* __restrict__ Xv = (const float4*)Xrow;
    __builtin_prefetch(Xrow, 0, 1);             // global_prefetch_b8

    v8f acc[8];
#pragma unroll
    for (int t = 0; t < 8; ++t) acc[t] = (v8f){0,0,0,0,0,0,0,0};

    // A double buffer: k-step kk uses float4 indices {b, b+1, b+4, b+5}, b = kk*8 + half*2
    float4 c0 = Xv[half * 2], c1 = Xv[half * 2 + 1];
    float4 c2 = Xv[half * 2 + 4], c3 = Xv[half * 2 + 5];

#pragma unroll
    for (int kk = 0; kk < 4; ++kk) {
        float4 n0, n1, n2, n3;
        if (kk < 3) {
            const int nb = (kk + 1) * 8 + half * 2;
            n0 = Xv[nb]; n1 = Xv[nb + 1]; n2 = Xv[nb + 4]; n3 = Xv[nb + 5];
        }
        Frag a;
        a.h[0][0] = (_Float16)c0.x; a.h[0][1] = (_Float16)c0.y;
        a.h[0][2] = (_Float16)c0.z; a.h[0][3] = (_Float16)c0.w;
        a.h[0][4] = (_Float16)c1.x; a.h[0][5] = (_Float16)c1.y;
        a.h[0][6] = (_Float16)c1.z; a.h[0][7] = (_Float16)c1.w;
        a.h[1][0] = (_Float16)c2.x; a.h[1][1] = (_Float16)c2.y;
        a.h[1][2] = (_Float16)c2.z; a.h[1][3] = (_Float16)c2.w;
        a.h[1][4] = (_Float16)c3.x; a.h[1][5] = (_Float16)c3.y;
        a.h[1][6] = (_Float16)c3.z; a.h[1][7] = (_Float16)c3.w;

        const int k0  = kk * 32;
        const int kb1 = k0 + half * 8;
        const int kb2 = k0 + 16 + half * 8;
#pragma unroll
        for (int t = 0; t < 8; ++t) {
            const int o = t * 16 + m;
            Frag b;
            b.h[0] = *reinterpret_cast<const h8*>(&Wl[o * 136 + kb1]);
            b.h[1] = *reinterpret_cast<const h8*>(&Wl[o * 136 + kb2]);
            acc[t] = __builtin_amdgcn_wmma_f32_16x16x32_f16(
                         false, a.v, false, b.v, (short)0, acc[t], false, false);
        }
        c0 = n0; c1 = n1; c2 = n2; c3 = n3;
    }

    // epilogue: C/D layout -> lane (m, half): col = t*16+m, row = v + 8*half
#pragma unroll
    for (int t = 0; t < 8; ++t) {
        const int col = colbase + t * 16 + m;
        const float bv = bias ? bias[col] : 0.f;
#pragma unroll
        for (int v = 0; v < 8; ++v) {
            const int row = rowbase + v + half * 8;
            float y = acc[t][v] + bv;
            if (do_relu) y = fmaxf(y, 0.f);
            const size_t off = (size_t)row * ldY + col;
            if (R) y += R[off];
            Y[off] = y;
        }
    }
}

// ---------------- f32 -> f16 weight cast ----------------
__global__ void cast_f16_kernel(const float* __restrict__ X, _Float16* __restrict__ Y, int n)
{
    const int i = blockIdx.x * blockDim.x + threadIdx.x;
    if (i < n) Y[i] = (_Float16)X[i];
}

// ---------------- BFS distances (chain edges within each subgraph) ------------
__global__ void bfs_kernel(int* __restrict__ dist)
{
    const int s = blockIdx.x * blockDim.x + threadIdx.x;
    if (s >= SS) return;
    int d[KK];
    d[0] = 0;
#pragma unroll
    for (int k = 1; k < KK; ++k) d[k] = MAXD;
#pragma unroll
    for (int it = 0; it < KK; ++it)
#pragma unroll
        for (int k = 1; k < KK; ++k) d[k] = min(d[k], d[k - 1] + 1);
#pragma unroll
    for (int k = 0; k < KK; ++k) dist[s * KK + k] = min(d[k], MAXD);
}

// ---------------- initial node features ----------------
__global__ void init_h_kernel(const int* __restrict__ atom_ids,
                              const int* __restrict__ node_ids,
                              const int* __restrict__ dist,
                              const float* __restrict__ atom_tab,
                              const float* __restrict__ dist_tab,
                              const float* __restrict__ log_probs,
                              const float* __restrict__ logp_W,
                              const float* __restrict__ logp_b,
                              float* __restrict__ h)
{
    const int i = blockIdx.x, c = threadIdx.x;
    const int nid = node_ids[i];
    const float valid = (nid >= 0) ? 1.f : 0.f;
    const int cl = nid > 0 ? nid : 0;
    const int sb = i / KK;
    const float x  = atom_tab[(size_t)atom_ids[cl] * HH + c];
    const float dp = dist_tab[(size_t)dist[i] * HH + c];
    const float lp = fmaxf(log_probs[sb] * logp_W[c] + logp_b[c], 0.f);
    h[(size_t)i * HH + c] = (x + dp + lp) * valid;
}

__global__ void count_roots_kernel(const int* __restrict__ node_ids, float* __restrict__ cnt)
{
    const int s = blockIdx.x * blockDim.x + threadIdx.x;
    if (s >= SS) return;
    int rid = node_ids[s * KK];
    if (rid < 0) rid = 0;
    atomicAdd(&cnt[rid], 1.f);
}

// ---------------- per-layer helpers ----------------
__global__ void bond_emb_kernel(const float* __restrict__ bond_tab,
                                const float* __restrict__ We,
                                const float* __restrict__ bWe,
                                float* __restrict__ be)
{
    const int b = blockIdx.x, c = threadIdx.x;   // grid 8, block 128
    float s = bWe[c];
    for (int k = 0; k < HH; ++k) s += bond_tab[b * HH + k] * We[(size_t)c * HH + k];
    be[b * HH + c] = s;
}

__global__ void hpre_kernel(const float* __restrict__ h,
                            const int* __restrict__ bond_ids,
                            const float* __restrict__ be,
                            const float* __restrict__ epsp,
                            float* __restrict__ t1)
{
    const int i = blockIdx.x, c = threadIdx.x;
    const float e0 = epsp[0];
    const int k = i % KK;
    float agg = 0.f;
    if (k > 0) {
        const int e = (i / KK) * (KK - 1) + (k - 1);
        const int bid = bond_ids[e];
        agg = fmaxf(h[(size_t)(i - 1) * HH + c] + be[bid * HH + c], 0.f);
    }
    t1[(size_t)i * HH + c] = (1.f + e0) * h[(size_t)i * HH + c] + agg;
}

__global__ void bn_stats_kernel(const float* __restrict__ X, int nrows, float* __restrict__ s)
{
    const int c = threadIdx.x;                      // 128 threads
    const int rpb = (nrows + gridDim.x - 1) / gridDim.x;
    const int r0 = blockIdx.x * rpb;
    const int r1 = min(r0 + rpb, nrows);
    float su = 0.f, sq = 0.f;
    for (int r = r0; r < r1; ++r) {
        const float x = X[(size_t)r * HH + c];
        su += x; sq += x * x;
    }
    atomicAdd(&s[c], su);
    atomicAdd(&s[HH + c], sq);
}

__global__ void hcan_acc_kernel(const float* __restrict__ h,
                                const int* __restrict__ node_ids,
                                float* __restrict__ hc)
{
    const int s = blockIdx.x, c = threadIdx.x;
    int rid = node_ids[s * KK];
    if (rid < 0) rid = 0;
    atomicAdd(&hc[(size_t)rid * HH + c], h[(size_t)(s * KK) * HH + c]);
}

__global__ void hcan_div_kernel(float* __restrict__ hc, const float* __restrict__ cnt)
{
    const int n = blockIdx.x, c = threadIdx.x;
    hc[(size_t)n * HH + c] /= fmaxf(cnt[n], 1.f);
}

__global__ void ginit_kernel(const float* __restrict__ hc, const float* __restrict__ epsp,
                             float* __restrict__ g)
{
    const int n = blockIdx.x, c = threadIdx.x;
    g[(size_t)n * HH + c] = (1.f + epsp[1]) * hc[(size_t)n * HH + c];
}

__global__ void edge_agg_kernel(const float* __restrict__ hc,
                                const int* __restrict__ ei,
                                const int* __restrict__ bids,
                                const float* __restrict__ be,
                                float* __restrict__ g)
{
    const int e = blockIdx.x, c = threadIdx.x;
    const int src = ei[e], dst = ei[ECANON + e];
    const float m = fmaxf(hc[(size_t)src * HH + c] + be[bids[e] * HH + c], 0.f);
    atomicAdd(&g[(size_t)dst * HH + c], m);
}

__global__ void bn_apply_kernel(const float* __restrict__ X, float invn,
                                const float* __restrict__ s,
                                const float* __restrict__ gw, const float* __restrict__ bw,
                                float* __restrict__ Y)
{
    const int n = blockIdx.x, c = threadIdx.x;
    const float mu = s[c] * invn;
    const float var = s[HH + c] * invn - mu * mu;
    const float rs = rsqrtf(var + 1e-5f);
    Y[(size_t)n * HH + c] = (X[(size_t)n * HH + c] - mu) * rs * gw[c] + bw[c];
}

__global__ void combine_kernel(const float* __restrict__ t1, const float* __restrict__ t2,
                               const float* __restrict__ hr4, const float* __restrict__ h_int,
                               const int* __restrict__ node_ids,
                               const float* __restrict__ s0,
                               const float* __restrict__ g0, const float* __restrict__ b0,
                               float* __restrict__ h)
{
    const int i = blockIdx.x, c = threadIdx.x;
    const int nid = node_ids[i];
    const float valid = (nid >= 0) ? 1.f : 0.f;
    const int cl = nid > 0 ? nid : 0;
    const float invn = 1.f / (float)SKK;
    const float mu = s0[c] * invn;
    const float var = s0[HH + c] * invn - mu * mu;
    const float rs = rsqrtf(var + 1e-5f);
    const float h1 = ((t1[(size_t)i * HH + c] - mu) * rs * g0[c] + b0[c]) * valid;
    float val;
    if ((i % KK) == 0)
        val = h1 + h_int[(size_t)cl * HH + c] * valid;
    else
        val = h1 + t2[(size_t)i * HH + c] + hr4[(size_t)(i / KK) * HH + c];
    h[(size_t)i * HH + c] = fmaxf(val, 0.f) * valid;
}

// ---------------- pooling / MHA ----------------
__global__ void hsub_kernel(const float* __restrict__ h, float* __restrict__ hs)
{
    const int s = blockIdx.x, c = threadIdx.x;
    float a = 0.f;
#pragma unroll
    for (int k = 0; k < KK; ++k) a += h[(size_t)(s * KK + k) * HH + c];
    hs[(size_t)s * HH + c] = a;
}

__global__ void perm_kernel(const int* __restrict__ node_ids, int* __restrict__ ctr,
                            int* __restrict__ perm)
{
    const int s = blockIdx.x * blockDim.x + threadIdx.x;
    if (s >= SS) return;
    int rid = node_ids[s * KK];
    if (rid < 0) rid = 0;
    const int slot = atomicAdd(&ctr[rid], 1);
    perm[s] = rid * MM + slot;
}

__global__ void scatter_kernel(const float* __restrict__ hs, const int* __restrict__ perm,
                               float* __restrict__ h2)
{
    const int s = blockIdx.x, c = threadIdx.x;
    h2[(size_t)perm[s] * HH + c] = hs[(size_t)s * HH + c];
}

__global__ void attn_kernel(const float* __restrict__ qkv, float* __restrict__ o)
{
    __shared__ float sq[4][4][32], skk[4][4][32], sv[4][4][32], ss[4][4][4];
    const int n = blockIdx.x, tid = threadIdx.x;
    const int hh = tid >> 5, d = tid & 31;
#pragma unroll
    for (int mI = 0; mI < 4; ++mI) {
        const size_t base = (size_t)(n * MM + mI) * 384;
        sq[mI][hh][d]  = qkv[base + hh * 32 + d];
        skk[mI][hh][d] = qkv[base + 128 + hh * 32 + d];
        sv[mI][hh][d]  = qkv[base + 256 + hh * 32 + d];
    }
    __syncthreads();
    if (tid < 64) {
        const int hd = tid >> 4, i = (tid >> 2) & 3, j = tid & 3;
        float s = 0.f;
        for (int dd = 0; dd < 32; ++dd) s += sq[i][hd][dd] * skk[j][hd][dd];
        ss[hd][i][j] = s * 0.17677669529663687f;    // 1/sqrt(32)
    }
    __syncthreads();
    if (tid < 16) {
        const int hd = tid >> 2, i = tid & 3;
        float mx = ss[hd][i][0];
        for (int j = 1; j < 4; ++j) mx = fmaxf(mx, ss[hd][i][j]);
        float e[4], sum = 0.f;
        for (int j = 0; j < 4; ++j) { e[j] = __expf(ss[hd][i][j] - mx); sum += e[j]; }
        for (int j = 0; j < 4; ++j) ss[hd][i][j] = e[j] / sum;
    }
    __syncthreads();
#pragma unroll
    for (int i = 0; i < 4; ++i) {
        float a = 0.f;
        for (int j = 0; j < 4; ++j) a += ss[hh][i][j] * sv[j][hh][d];
        o[(size_t)(n * MM + i) * HH + hh * 32 + d] = a;
    }
}

__global__ void mean_kernel(const float* __restrict__ ha, float* __restrict__ ne)
{
    const int n = blockIdx.x, c = threadIdx.x;
    float a = 0.f;
#pragma unroll
    for (int mI = 0; mI < 4; ++mI) a += ha[(size_t)(n * MM + mI) * HH + c];
    ne[(size_t)n * HH + c] = a * 0.25f;
}

__global__ void out_kernel(const float* __restrict__ ne, const float* __restrict__ s2,
                           const float* __restrict__ rg, const float* __restrict__ rb,
                           const int* __restrict__ batch, float* __restrict__ out)
{
    const int n = blockIdx.x, c = threadIdx.x;
    const float invn = 1.f / (float)NT;
    const float mu = s2[c] * invn;
    const float var = s2[HH + c] * invn - mu * mu;
    const float rs = rsqrtf(var + 1e-5f);
    const float v = (ne[(size_t)n * HH + c] - mu) * rs * rg[c] + rb[c];
    atomicAdd(&out[(size_t)batch[n] * HH + c], v);
}

// =====================================================================
extern "C" void kernel_launch(void* const* d_in, const int* in_sizes, int n_in,
                              void* d_out, int out_size, void* d_ws, size_t ws_size,
                              hipStream_t stream)
{
    (void)in_sizes; (void)n_in; (void)out_size; (void)ws_size;

    const int*   atom_ids       = (const int*)  d_in[0];
    const int*   node_ids       = (const int*)  d_in[1];
    // d_in[2] = intra_ei (chain structure used implicitly)
    const int*   intra_bond_ids = (const int*)  d_in[3];
    const int*   edge_index     = (const int*)  d_in[4];
    const int*   canon_bond_ids = (const int*)  d_in[5];
    const int*   batch          = (const int*)  d_in[6];
    const float* log_probs      = (const float*)d_in[7];
    const float* atom_tab       = (const float*)d_in[8];
    const float* bond_tab       = (const float*)d_in[9];
    const float* dist_tab       = (const float*)d_in[10];
    const float* logp_W         = (const float*)d_in[11];
    const float* logp_b         = (const float*)d_in[12];
    const float* lw             = (const float*)d_in[13];
    const float* lb             = (const float*)d_in[14];
    const float* bn_g           = (const float*)d_in[15];
    const float* bn_b           = (const float*)d_in[16];
    const float* epsv           = (const float*)d_in[17];
    const float* mha_in_W       = (const float*)d_in[18];
    const float* mha_in_b       = (const float*)d_in[19];
    const float* mha_out_W      = (const float*)d_in[20];
    const float* mha_out_b      = (const float*)d_in[21];
    const float* ro_g           = (const float*)d_in[22];
    const float* ro_b           = (const float*)d_in[23];

    // ---- workspace carve-up (float element offsets) ----
    float* wsf = (float*)d_ws;
    const size_t BIG = (size_t)SKK * HH;           // 25,165,824
    float* h    = wsf;
    float* t1   = wsf + BIG;
    float* t2   = wsf + 2 * BIG;
    float* hr4  = wsf + 3 * BIG;                   // S*H
    float* hc   = hr4 + (size_t)SS * HH;
    float* g    = hc  + (size_t)NT * HH;
    float* ti1  = g   + (size_t)NT * HH;
    float* ti2  = ti1 + (size_t)NT * HH;
    float* hint = ti2 + (size_t)NT * HH;
    float* be_i = hint + (size_t)NT * HH;          // 8*H
    float* be_c = be_i + 8 * HH;
    float* cnt  = be_c + 8 * HH;                   // NT floats
    float* s0   = cnt + NT;
    float* s1   = s0 + 2 * HH;
    float* s2   = s1 + 2 * HH;
    int*   dist = (int*)(s2 + 2 * HH);             // SK ints
    int*   ctr  = dist + SKK;                      // NT ints
    int*   perm = ctr + NT;                        // S ints
    // f16 weight arena (after int region; 16B-aligned by construction)
    _Float16* lwh       = (_Float16*)(perm + SS);                    // L*8*H*H halves
    _Float16* mha_in_h  = lwh + (size_t)LL * 8 * HH * HH;            // 384*H
    _Float16* mha_out_h = mha_in_h + (size_t)384 * HH;               // H*H
    // post-layer overlays (inside t1/t2 which are free after the layer loop)
    float* h_sub  = t1;
    float* h2buf  = t1 + (size_t)SS * HH;
    float* ne     = h2buf + (size_t)SS * HH;
    float* qkv    = t2;
    float* attn_o = t2 + (size_t)SS * 384;
    float* ha     = attn_o + (size_t)SS * HH;

    auto gemm = [&](const float* X, int xstep, const _Float16* W, const float* bias,
                    const float* R, float* Y, int ldY, int relu, int nrows, int oc) {
        dim3 gdim(nrows / 128, oc / 128);
        gemm_wmma_kernel<<<gdim, 256, 0, stream>>>(X, xstep, W, bias, R, Y, ldY, relu);
    };

    // ---- setup ----
    cast_f16_kernel<<<(LL * 8 * HH * HH) / 256, 256, 0, stream>>>(lw, lwh, LL * 8 * HH * HH);
    cast_f16_kernel<<<(384 * HH) / 256, 256, 0, stream>>>(mha_in_W, mha_in_h, 384 * HH);
    cast_f16_kernel<<<(HH * HH) / 256, 256, 0, stream>>>(mha_out_W, mha_out_h, HH * HH);

    hipMemsetAsync(cnt, 0, NT * sizeof(float), stream);
    count_roots_kernel<<<SS / 256, 256, 0, stream>>>(node_ids, cnt);
    bfs_kernel<<<SS / 256, 256, 0, stream>>>(dist);
    init_h_kernel<<<SKK, 128, 0, stream>>>(atom_ids, node_ids, dist, atom_tab,
                                           dist_tab, log_probs, logp_W, logp_b, h);

    // ---- GNN layers ----
    for (int l = 0; l < LL; ++l) {
        const float*    Wl_ = lw  + (size_t)l * 8 * HH * HH;
        const _Float16* Wh_ = lwh + (size_t)l * 8 * HH * HH;
        const float*    bl_ = lb  + (size_t)l * 8 * HH;
        const float*    ep  = epsv + 2 * l;

        bond_emb_kernel<<<8, 128, 0, stream>>>(bond_tab, Wl_,             bl_,          be_i);
        bond_emb_kernel<<<8, 128, 0, stream>>>(bond_tab, Wl_ + 5*HH*HH,   bl_ + 5*HH,   be_c);

        hpre_kernel<<<SKK, 128, 0, stream>>>(h, intra_bond_ids, be_i, ep, t1);
        gemm(t1, 1, Wh_ + 1*HH*HH, bl_ + 1*HH, nullptr, t2, HH, 1, SKK, HH); // relu(h@W1+b1)
        gemm(t2, 1, Wh_ + 2*HH*HH, bl_ + 2*HH, nullptr, t1, HH, 0, SKK, HH); // @W2+b2

        hipMemsetAsync(s0, 0, 2 * HH * sizeof(float), stream);
        bn_stats_kernel<<<96, 128, 0, stream>>>(t1, SKK, s0);

        gemm(h, 1,  Wh_ + 3*HH*HH, bl_ + 3*HH, nullptr, t2,  HH, 0, SKK, HH); // h@W3+b3
        gemm(h, KK, Wh_ + 4*HH*HH, bl_ + 4*HH, nullptr, hr4, HH, 0, SS,  HH); // roots@W4+b4

        hipMemsetAsync(hc, 0, (size_t)NT * HH * sizeof(float), stream);
        hcan_acc_kernel<<<SS, 128, 0, stream>>>(h, node_ids, hc);
        hcan_div_kernel<<<NT, 128, 0, stream>>>(hc, cnt);

        ginit_kernel<<<NT, 128, 0, stream>>>(hc, ep, g);
        edge_agg_kernel<<<ECANON, 128, 0, stream>>>(hc, edge_index, canon_bond_ids, be_c, g);
        gemm(g,   1, Wh_ + 6*HH*HH, bl_ + 6*HH, nullptr, ti1, HH, 1, NT, HH);
        gemm(ti1, 1, Wh_ + 7*HH*HH, bl_ + 7*HH, nullptr, ti2, HH, 0, NT, HH);
        hipMemsetAsync(s1, 0, 2 * HH * sizeof(float), stream);
        bn_stats_kernel<<<32, 128, 0, stream>>>(ti2, NT, s1);
        bn_apply_kernel<<<NT, 128, 0, stream>>>(ti2, 1.f / NT, s1,
                                                bn_g + (size_t)(2*l+1)*HH,
                                                bn_b + (size_t)(2*l+1)*HH, hint);

        combine_kernel<<<SKK, 128, 0, stream>>>(t1, t2, hr4, hint, node_ids, s0,
                                                bn_g + (size_t)(2*l)*HH,
                                                bn_b + (size_t)(2*l)*HH, h);
    }

    // ---- subgraph pooling + reorder ----
    hsub_kernel<<<SS, 128, 0, stream>>>(h, h_sub);
    hipMemsetAsync(ctr, 0, NT * sizeof(int), stream);
    hipMemsetAsync(h2buf, 0, (size_t)SS * HH * sizeof(float), stream);
    perm_kernel<<<SS / 256, 256, 0, stream>>>(node_ids, ctr, perm);
    scatter_kernel<<<SS, 128, 0, stream>>>(h_sub, perm, h2buf);

    // ---- MHA ----
    gemm(h2buf, 1, mha_in_h, mha_in_b, nullptr, qkv, 384, 0, SS, 384);
    attn_kernel<<<NT, 128, 0, stream>>>(qkv, attn_o);
    gemm(attn_o, 1, mha_out_h, mha_out_b, h2buf, ha, HH, 0, SS, HH);
    mean_kernel<<<NT, 128, 0, stream>>>(ha, ne);

    // ---- readout BN + batch pooling ----
    hipMemsetAsync(s2, 0, 2 * HH * sizeof(float), stream);
    bn_stats_kernel<<<32, 128, 0, stream>>>(ne, NT, s2);
    hipMemsetAsync(d_out, 0, (size_t)BB * HH * sizeof(float), stream);
    out_kernel<<<NT, 128, 0, stream>>>(ne, s2, ro_g, ro_b, batch, (float*)d_out);
}
